// RetNetBlock_79027398246801
// MI455X (gfx1250) — compile-verified, matrix operand order checked by reading, not run on hardware
//
#include <hip/hip_runtime.h>
#include <hip/hip_bf16.h>

// ---------------------------------------------------------------------------
// RetNetBlock (linear attention form) for MI455X / gfx1250.
//   out = x + proj( (ln(x)Wqkv -> q,k,v ; q (k^T v)/sqrt(C)) )
// All GEMMs run on v_wmma_f32_16x16x32_bf16 (bf16 in, f32 accumulate).
// ---------------------------------------------------------------------------

typedef __bf16 v8bf  __attribute__((ext_vector_type(8)));
typedef __bf16 v16bf __attribute__((ext_vector_type(16)));
typedef float  v8f   __attribute__((ext_vector_type(8)));

#define LDSPAD 40   // 32 K-elements + 8 pad (80B row stride, 16B aligned)

// ---------------------------------------------------------------------------
// fp32 -> bf16 elementwise conversion (for weights)
// ---------------------------------------------------------------------------
__global__ void cvt_f32_bf16(const float* __restrict__ src,
                             __bf16* __restrict__ dst, int n) {
    int i = blockIdx.x * blockDim.x + threadIdx.x;
    if (i < n) dst[i] = (__bf16)src[i];
}

// ---------------------------------------------------------------------------
// LayerNorm: one 256-thread block per row of C=1024, bf16 output
// ---------------------------------------------------------------------------
__global__ __launch_bounds__(256)
void layernorm_bf16(const float* __restrict__ x,
                    const float* __restrict__ w,
                    const float* __restrict__ b,
                    __bf16* __restrict__ h, int C) {
    int row = blockIdx.x;
    const float* xr = x + (size_t)row * C;

    float s = 0.f, s2 = 0.f;
    for (int c = threadIdx.x; c < C; c += blockDim.x) {
        float v = xr[c];
        s += v; s2 += v * v;
    }
    // wave32 reduction
    #pragma unroll
    for (int o = 16; o > 0; o >>= 1) {
        s  += __shfl_down(s,  o, 32);
        s2 += __shfl_down(s2, o, 32);
    }
    __shared__ float rs[8], rs2[8];
    __shared__ float mu_s, rstd_s;
    int wid = threadIdx.x >> 5;
    if ((threadIdx.x & 31) == 0) { rs[wid] = s; rs2[wid] = s2; }
    __syncthreads();
    if (threadIdx.x == 0) {
        float ts = 0.f, ts2 = 0.f;
        #pragma unroll
        for (int i = 0; i < 8; ++i) { ts += rs[i]; ts2 += rs2[i]; }
        float mu  = ts / (float)C;
        float var = ts2 / (float)C - mu * mu;
        mu_s = mu;
        rstd_s = rsqrtf(var + 1e-5f);
    }
    __syncthreads();
    float mu = mu_s, r = rstd_s;
    for (int c = threadIdx.x; c < C; c += blockDim.x) {
        h[(size_t)row * C + c] = (__bf16)((xr[c] - mu) * r * w[c] + b[c]);
    }
}

// ---------------------------------------------------------------------------
// Tiled WMMA GEMM:  D(M,N) = A(M,K) @ W(N,K)^T   (all bf16, f32 accumulate)
//   block tile 128x128, 8 waves, wave tile 32x64 (2x4 wmma 16x16 tiles)
//   K step 32, LDS-staged operands, ds_load_b128 fragment reads.
// Epilogue modes:
//   0: qkv split  -> q (bias, *1/sqrt(C)), kT/vT (bias, [B,C,T] transposed)
//   1: store bf16 -> outb[m*N+n]
//   2: fp32 out   -> outf[m*N+n] = acc + bias[n] + residual[m*N+n]
// ---------------------------------------------------------------------------
__global__ __launch_bounds__(256)
void gemm_wmma_bf16(const __bf16* __restrict__ Abase,
                    const __bf16* __restrict__ Wbase,
                    __bf16* outb, float* outf,
                    const float* __restrict__ residual,
                    const float* __restrict__ bias,
                    __bf16* qout, __bf16* kTout, __bf16* vTout,
                    int M, int N, int K, int mode, int Cdim, int Tdim,
                    unsigned long long aStride, unsigned long long wStride,
                    unsigned long long oStride) {
    __shared__ __bf16 As[128 * LDSPAD];
    __shared__ __bf16 Bs[128 * LDSPAD];

    const __bf16* A = Abase + (size_t)blockIdx.z * aStride;
    const __bf16* W = Wbase + (size_t)blockIdx.z * wStride;

    const int tid   = threadIdx.x;
    const int lane  = tid & 31;
    const int wave  = tid >> 5;
    const int waveM = wave & 3;   // 0..3 -> 32-row strip
    const int waveN = wave >> 2;  // 0..1 -> 64-col strip
    const int mBlock = blockIdx.x * 128;
    const int nBlock = blockIdx.y * 128;

    const int col  = lane & 15;   // N (or M) index inside a 16x16 tile
    const int hsel = lane >> 4;   // which K half-pattern this lane holds

    v8f acc[2][4];
    {
        v8f z = {0.f, 0.f, 0.f, 0.f, 0.f, 0.f, 0.f, 0.f};
        #pragma unroll
        for (int mi = 0; mi < 2; ++mi)
            #pragma unroll
            for (int ni = 0; ni < 4; ++ni)
                acc[mi][ni] = z;
    }

    for (int k0 = 0; k0 < K; k0 += 32) {
        // ---- cooperative global -> LDS stage (2 x b128 per thread per tile)
        #pragma unroll
        for (int i = 0; i < 2; ++i) {
            int chunk = tid + (i << 8);       // 0..511
            int row   = chunk >> 2;           // 0..127
            int c8    = (chunk & 3) << 3;     // 0,8,16,24
            *reinterpret_cast<v8bf*>(&As[row * LDSPAD + c8]) =
                *reinterpret_cast<const v8bf*>(A + (size_t)(mBlock + row) * K + k0 + c8);
            *reinterpret_cast<v8bf*>(&Bs[row * LDSPAD + c8]) =
                *reinterpret_cast<const v8bf*>(W + (size_t)(nBlock + row) * K + k0 + c8);
        }
        // ---- prefetch next K tile into caches (global_prefetch)
        if (k0 + 32 < K) {
            int row = tid >> 2;
            int c8  = (tid & 3) << 3;
            __builtin_prefetch(A + (size_t)(mBlock + row) * K + k0 + 32 + c8, 0, 1);
            __builtin_prefetch(W + (size_t)(nBlock + row) * K + k0 + 32 + c8, 0, 1);
        }
        __syncthreads();

        // ---- fragment loads from LDS (16-bit A-layout; symmetric for B)
        v16bf af[2], bfr[4];
        #pragma unroll
        for (int mi = 0; mi < 2; ++mi) {
            int r = waveM * 32 + mi * 16 + col;
            v8bf lo = *reinterpret_cast<const v8bf*>(&As[r * LDSPAD + hsel * 8]);
            v8bf hi = *reinterpret_cast<const v8bf*>(&As[r * LDSPAD + 16 + hsel * 8]);
            af[mi] = __builtin_shufflevector(lo, hi, 0,1,2,3,4,5,6,7,8,9,10,11,12,13,14,15);
        }
        #pragma unroll
        for (int ni = 0; ni < 4; ++ni) {
            int r = waveN * 64 + ni * 16 + col;
            v8bf lo = *reinterpret_cast<const v8bf*>(&Bs[r * LDSPAD + hsel * 8]);
            v8bf hi = *reinterpret_cast<const v8bf*>(&Bs[r * LDSPAD + 16 + hsel * 8]);
            bfr[ni] = __builtin_shufflevector(lo, hi, 0,1,2,3,4,5,6,7,8,9,10,11,12,13,14,15);
        }

        // ---- 8x v_wmma_f32_16x16x32_bf16
        #pragma unroll
        for (int mi = 0; mi < 2; ++mi)
            #pragma unroll
            for (int ni = 0; ni < 4; ++ni)
                acc[mi][ni] = __builtin_amdgcn_wmma_f32_16x16x32_bf16(
                    false, af[mi], false, bfr[ni],
                    (short)0, acc[mi][ni], false, false);

        __syncthreads();
    }

    // ---- epilogue: lane holds column `col`, rows hsel*8 .. hsel*8+7
    const int rbase = hsel * 8;
    const float qscale = rsqrtf((float)Cdim);
    #pragma unroll
    for (int mi = 0; mi < 2; ++mi) {
        #pragma unroll
        for (int ni = 0; ni < 4; ++ni) {
            int gn = nBlock + waveN * 64 + ni * 16 + col;
            #pragma unroll
            for (int r = 0; r < 8; ++r) {
                int gm = mBlock + waveM * 32 + mi * 16 + rbase + r;
                float v = acc[mi][ni][r];
                if (mode == 0) {
                    float bb = bias[gn];
                    int bi = gm / Tdim;
                    int t  = gm - bi * Tdim;
                    if (gn < Cdim) {
                        qout[(size_t)gm * Cdim + gn] = (__bf16)((v + bb) * qscale);
                    } else if (gn < 2 * Cdim) {
                        int c = gn - Cdim;
                        kTout[((size_t)bi * Cdim + c) * Tdim + t] = (__bf16)(v + bb);
                    } else {
                        int c = gn - 2 * Cdim;
                        vTout[((size_t)bi * Cdim + c) * Tdim + t] = (__bf16)(v + bb);
                    }
                } else if (mode == 1) {
                    __bf16* OB = outb + (size_t)blockIdx.z * oStride;
                    OB[(size_t)gm * N + gn] = (__bf16)v;
                } else {
                    size_t idx = (size_t)gm * N + gn;
                    outf[idx] = v + bias[gn] + residual[idx];
                }
            }
        }
    }
}

// ---------------------------------------------------------------------------
// Host orchestration
// ---------------------------------------------------------------------------
extern "C" void kernel_launch(void* const* d_in, const int* in_sizes, int n_in,
                              void* d_out, int out_size, void* d_ws, size_t ws_size,
                              hipStream_t stream) {
    (void)in_sizes; (void)n_in; (void)out_size; (void)ws_size;

    const float* x      = (const float*)d_in[0];
    const float* norm_w = (const float*)d_in[1];
    const float* norm_b = (const float*)d_in[2];
    const float* qkv_w  = (const float*)d_in[3];
    const float* qkv_b  = (const float*)d_in[4];
    const float* proj_w = (const float*)d_in[5];
    const float* proj_b = (const float*)d_in[6];
    float* out = (float*)d_out;

    const int Bn = 4, T = 4096, C = 1024;
    const int BT = Bn * T;                 // 16384

    // workspace carve-up (bf16 buffers)
    char* ws = (char*)d_ws;
    size_t off = 0;
    auto carve = [&](size_t bytes) -> void* {
        void* p = ws + off;
        off += (bytes + 255) & ~(size_t)255;
        return p;
    };
    __bf16* h   = (__bf16*)carve((size_t)BT * C * 2);       // ln output
    __bf16* qwb = (__bf16*)carve((size_t)3 * C * C * 2);    // qkv_w bf16
    __bf16* pwb = (__bf16*)carve((size_t)C * C * 2);        // proj_w bf16
    __bf16* q   = (__bf16*)carve((size_t)BT * C * 2);       // q (pre-scaled)
    __bf16* kT  = (__bf16*)carve((size_t)BT * C * 2);       // k^T  [B,C,T]
    __bf16* vT  = (__bf16*)carve((size_t)BT * C * 2);       // v^T  [B,C,T]
    __bf16* Mt  = (__bf16*)carve((size_t)Bn * C * C * 2);   // (k^T v)^T per batch
    __bf16* o   = (__bf16*)carve((size_t)BT * C * 2);       // attention output

    // 1) weight conversion
    cvt_f32_bf16<<<(3 * C * C + 255) / 256, 256, 0, stream>>>(qkv_w, qwb, 3 * C * C);
    cvt_f32_bf16<<<(C * C + 255) / 256, 256, 0, stream>>>(proj_w, pwb, C * C);

    // 2) layernorm
    layernorm_bf16<<<BT, 256, 0, stream>>>(x, norm_w, norm_b, h, C);

    // 3) qkv = h @ qkv_w^T + b ; split into q (scaled), kT, vT
    gemm_wmma_bf16<<<dim3(BT / 128, (3 * C) / 128, 1), 256, 0, stream>>>(
        h, qwb, nullptr, nullptr, nullptr, qkv_b, q, kT, vT,
        BT, 3 * C, C, /*mode=*/0, C, T, 0ULL, 0ULL, 0ULL);

    // 4) Mt[b] = vT[b] @ kT[b]^T  == (k^T v)^T per batch   (K = T)
    gemm_wmma_bf16<<<dim3(C / 128, C / 128, Bn), 256, 0, stream>>>(
        vT, kT, Mt, nullptr, nullptr, nullptr, nullptr, nullptr, nullptr,
        C, C, T, /*mode=*/1, C, T,
        (unsigned long long)C * T, (unsigned long long)C * T,
        (unsigned long long)C * C);

    // 5) o[b] = q[b] @ Mt[b]^T  == q (k^T v) / sqrt(C)
    gemm_wmma_bf16<<<dim3(T / 128, C / 128, Bn), 256, 0, stream>>>(
        q, Mt, o, nullptr, nullptr, nullptr, nullptr, nullptr, nullptr,
        T, C, C, /*mode=*/1, C, T,
        (unsigned long long)T * C, (unsigned long long)C * C,
        (unsigned long long)T * C);

    // 6) out = o @ proj_w^T + proj_b + x
    gemm_wmma_bf16<<<dim3(BT / 128, C / 128, 1), 256, 0, stream>>>(
        o, pwb, nullptr, out, x, proj_b, nullptr, nullptr, nullptr,
        BT, C, C, /*mode=*/2, C, T, 0ULL, 0ULL, 0ULL);
}
